// Attention_72773925863713
// MI455X (gfx1250) — compile-verified, matrix operand order checked by reading, not run on hardware
//
#include <hip/hip_runtime.h>
#include <stdint.h>
#include <stddef.h>

// Problem constants (match reference)
#define B_   2
#define S_   2048
#define D_   1024
#define H_   16
#define HD_  64
#define D3_  3072
#define MASKED_BIAS (-10000.0f)

typedef __bf16 bf16_t;
typedef bf16_t v16bf  __attribute__((ext_vector_type(16)));
typedef bf16_t bf16x2 __attribute__((ext_vector_type(2)));
typedef float  f32x8  __attribute__((ext_vector_type(8)));

// ---------------------------------------------------------------------------
// WMMA helpers (CDNA5 v_wmma_f32_16x16x32_bf16, wave32)
// ---------------------------------------------------------------------------
__device__ __forceinline__ f32x8 wmma_bf16(v16bf a, v16bf b, f32x8 c) {
  return __builtin_amdgcn_wmma_f32_16x16x32_bf16(false, a, false, b,
                                                 (short)0, c, false, false);
}

// K offset of bf16 pair j (0..7) within a 32-wide K slab, for lane-half `half`
__device__ __forceinline__ int kOff(int j, int half) {
  return (j < 4) ? (2 * j + 8 * half) : (16 + 2 * (j - 4) + 8 * half);
}

// 16-element bf16 fragment from row-major [*, ld] storage (A rows / B^T rows).
__device__ __forceinline__ v16bf load_frag_bf16(const bf16_t* __restrict__ base,
                                                int row, int ld, int k0, int half) {
  v16bf f;
  const bf16_t* p = base + (size_t)row * ld + k0;
#pragma unroll
  for (int j = 0; j < 8; ++j) {
    bf16x2 w = *reinterpret_cast<const bf16x2*>(p + kOff(j, half));
    f[2 * j]     = w[0];
    f[2 * j + 1] = w[1];
  }
  return f;
}

// Fragment from an f32 row in LDS (softmax probabilities), cvt -> bf16.
__device__ __forceinline__ v16bf load_frag_lds_f32(const float* p, int k0, int half) {
  v16bf f;
#pragma unroll
  for (int j = 0; j < 8; ++j) {
    float2 w = *reinterpret_cast<const float2*>(p + k0 + kOff(j, half));
    f[2 * j]     = (bf16_t)w.x;
    f[2 * j + 1] = (bf16_t)w.y;
  }
  return f;
}

// ---------------------------------------------------------------------------
// Kernel 1a: f32 -> bf16 copy (hidden_states)
// ---------------------------------------------------------------------------
__global__ __launch_bounds__(256) void k_cvt_bf16(bf16_t* __restrict__ dst,
                                                  const float* __restrict__ src,
                                                  int n) {
  for (int i = blockIdx.x * blockDim.x + threadIdx.x; i < n;
       i += gridDim.x * blockDim.x)
    dst[i] = (bf16_t)src[i];
}

// Kernel 1b: dst[n*K + k] = (bf16) src[k*N + n]  (weight transpose+convert)
__global__ __launch_bounds__(256) void k_transpose_cvt(bf16_t* __restrict__ dst,
                                                       const float* __restrict__ src,
                                                       int K, int N) {
  size_t total = (size_t)K * N;
  for (size_t i = blockIdx.x * (size_t)blockDim.x + threadIdx.x; i < total;
       i += (size_t)gridDim.x * blockDim.x) {
    int n = (int)(i / K);
    int k = (int)(i % K);
    dst[i] = (bf16_t)src[(size_t)k * N + n];
  }
}

// ---------------------------------------------------------------------------
// Kernel 2: QKV GEMM, 64x64 per wave (4x4 WMMA register blocking).
// Scatter to Q/K bf16 [bh,s,hd] and V^T bf16 [bh,hd,s].
// ---------------------------------------------------------------------------
__global__ __launch_bounds__(256) void k_gemm_qkv(const bf16_t* __restrict__ X,
                                                  const bf16_t* __restrict__ WaT,
                                                  const float* __restrict__ b_attn,
                                                  bf16_t* __restrict__ Qb,
                                                  bf16_t* __restrict__ Kb,
                                                  bf16_t* __restrict__ Vt) {
  const int lane = threadIdx.x & 31;
  const int wave = threadIdx.x >> 5;
  const int NT64 = D3_ / 64;                         // 48
  const int total = (B_ * S_ / 64) * NT64;           // 64*48 = 3072 waves
  int tile = blockIdx.x * 8 + wave;
  if (tile >= total) return;
  const int m0 = (tile / NT64) * 64;
  const int n0 = (tile % NT64) * 64;
  const int half = lane >> 4, l16 = lane & 15;

  f32x8 c[4][4] = {};
  for (int k0 = 0; k0 < D_; k0 += 32) {
    v16bf a[4], b[4];
#pragma unroll
    for (int i = 0; i < 4; ++i)
      a[i] = load_frag_bf16(X, m0 + i * 16 + l16, D_, k0, half);
#pragma unroll
    for (int i = 0; i < 4; ++i)
      b[i] = load_frag_bf16(WaT, n0 + i * 16 + l16, D_, k0, half);
#pragma unroll
    for (int i = 0; i < 4; ++i)
#pragma unroll
      for (int jj = 0; jj < 4; ++jj)
        c[i][jj] = wmma_bf16(a[i], b[jj], c[i][jj]);
  }

#pragma unroll
  for (int jj = 0; jj < 4; ++jj) {
    const int gn = n0 + jj * 16 + l16;               // col in [3D)
    const float bias = b_attn[gn];
    const int sec = gn >> 10;                        // 0=Q 1=K 2=V
    const int dD  = gn & (D_ - 1);
    const int h   = dD >> 6, hd = dD & (HD_ - 1);
#pragma unroll
    for (int i = 0; i < 4; ++i) {
#pragma unroll
      for (int j = 0; j < 8; ++j) {
        int gm = m0 + i * 16 + j + 8 * half;         // row in [B*S)
        float val = c[i][jj][j] + bias;
        int bI = gm / S_, sI = gm % S_;
        size_t bh = (size_t)bI * H_ + h;
        if (sec == 0)
          Qb[(bh * S_ + sI) * HD_ + hd] = (bf16_t)val;
        else if (sec == 1)
          Kb[(bh * S_ + sI) * HD_ + hd] = (bf16_t)val;
        else
          Vt[(bh * HD_ + hd) * (size_t)S_ + sI] = (bf16_t)val;
      }
    }
  }
}

// ---------------------------------------------------------------------------
// Kernel 3: FUSED attention core. One workgroup (256 thr = 8 waves) per
// (batch*head, 16-query tile). Scores for the causal prefix are computed with
// WMMA into a 16x2048 f32 LDS buffer (128 KB -- CDNA5 320KB/WGP LDS), softmax
// is done in LDS, probabilities are written once to d_out (mandatory output),
// and P @ V runs straight out of LDS so the 537MB array never re-crosses HBM.
// ---------------------------------------------------------------------------
__global__ __launch_bounds__(256) void k_attn(const bf16_t* __restrict__ Qb,
                                              const bf16_t* __restrict__ Kb,
                                              const bf16_t* __restrict__ Vt,
                                              const float* __restrict__ amask,
                                              float* __restrict__ w,
                                              bf16_t* __restrict__ Y) {
  __shared__ float Ssc[16 * S_];        // 128 KB scores -> probabilities
  __shared__ float red16[16 * 16];      // per-row reduction scratch
  __shared__ float accv[2 * 16 * 64];   // P@V partial sums (2 K-halves)

  const int bh = blockIdx.x >> 7;       // 0..31
  const int qt = blockIdx.x & 127;      // 0..127
  const int bI = bh / H_, hI = bh % H_;
  const int t = threadIdx.x, wave = t >> 5, lane = t & 31;
  const int half = lane >> 4, l16 = lane & 15;

  const int ktLimit = qt + 1;           // causal: key tiles with live columns
  const int kmax = ktLimit * 16;        // live key count (multiple of 16)
  const int kpad = (kmax + 31) & ~31;   // padded to a 32-wide WMMA slab

  // ---- Phase 1: scores = QK^T/8 + masks, distributed over 8 waves ----
  const bf16_t* qbase = Qb + (size_t)bh * S_ * HD_;
  const bf16_t* kbase = Kb + (size_t)bh * S_ * HD_;
  const int m = qt * 16 + l16;
  const v16bf qa0 = load_frag_bf16(qbase, m, HD_, 0, half);
  const v16bf qa1 = load_frag_bf16(qbase, m, HD_, 32, half);

  for (int kt = wave; kt < ktLimit; kt += 8) {
    const int n = kt * 16 + l16;
    f32x8 c = {};
    c = wmma_bf16(qa0, load_frag_bf16(kbase, n, HD_, 0, half), c);
    c = wmma_bf16(qa1, load_frag_bf16(kbase, n, HD_, 32, half), c);
    const float am = amask[(size_t)bI * S_ + n];
#pragma unroll
    for (int j = 0; j < 8; ++j) {
      int r  = j + 8 * half;            // row within q-tile
      int gm = qt * 16 + r;
      float val = c[j] * 0.125f;        // 1/sqrt(64)
      val = (n > gm) ? MASKED_BIAS : val;
      Ssc[r * S_ + n] = val + am;
    }
  }
  __syncthreads();

  // ---- Phase 2: row softmax in LDS; write probs to d_out once ----
  const int row = t >> 4;               // 16 threads per query row
  const int idx = t & 15;
  float* srow = Ssc + row * S_;

  float mx = -3.0e38f;
  for (int c0 = idx; c0 < kmax; c0 += 16) mx = fmaxf(mx, srow[c0]);
  red16[row * 16 + idx] = mx;
  __syncthreads();
  for (int off = 8; off > 0; off >>= 1) {
    if (idx < off)
      red16[row * 16 + idx] = fmaxf(red16[row * 16 + idx], red16[row * 16 + idx + off]);
    __syncthreads();
  }
  mx = red16[row * 16];
  __syncthreads();

  float sum = 0.f;
  for (int c0 = idx; c0 < kmax; c0 += 16) {
    float e = __expf(srow[c0] - mx);
    srow[c0] = e;
    sum += e;
  }
  red16[row * 16 + idx] = sum;
  __syncthreads();
  for (int off = 8; off > 0; off >>= 1) {
    if (idx < off)
      red16[row * 16 + idx] += red16[row * 16 + idx + off];
    __syncthreads();
  }
  const float inv = 1.0f / red16[row * 16];

  float* wrow = w + ((size_t)bh * S_ + qt * 16 + row) * S_;
  for (int c0 = idx; c0 < kmax; c0 += 16) {
    float p = srow[c0] * inv;
    srow[c0] = p;                       // probs stay in LDS for P@V
    wrow[c0] = p;                       // mandatory attn_weights output
  }
  for (int c0 = kmax + idx; c0 < kpad; c0 += 16) srow[c0] = 0.f;  // WMMA pad
  for (int c0 = kmax + idx; c0 < S_; c0 += 16) wrow[c0] = 0.f;    // exact zeros
  __syncthreads();

  // ---- Phase 3: P @ V from LDS. 8 waves = 4 hd-tiles x 2 K-halves ----
  const int dt = wave >> 1;             // 0..3 (16-wide hd tile)
  const int kh = wave & 1;              // K-split
  const bf16_t* vbase = Vt + (size_t)bh * HD_ * S_;
  const int n = dt * 16 + l16;
  const int kSteps = kpad >> 5;
  const float* arow = Ssc + l16 * S_;   // A row = query l16

  f32x8 c = {};
  for (int ks = kh; ks < kSteps; ks += 2) {
    v16bf a = load_frag_lds_f32(arow, ks * 32, half);
    v16bf b = load_frag_bf16(vbase, n, S_, ks * 32, half);
    c = wmma_bf16(a, b, c);
  }
#pragma unroll
  for (int j = 0; j < 8; ++j)
    accv[(kh * 16 + j + 8 * half) * 64 + dt * 16 + l16] = c[j];
  __syncthreads();

  // combine K-halves, emit bf16 merged-head activations
  for (int i = t; i < 16 * 64; i += 256) {
    int r = i >> 6, ccol = i & 63;
    float v = accv[r * 64 + ccol] + accv[(16 + r) * 64 + ccol];
    int q = qt * 16 + r;
    Y[((size_t)bI * S_ + q) * D_ + hI * HD_ + ccol] = (bf16_t)v;
  }
}

// ---------------------------------------------------------------------------
// Kernel 4: output projection, 64x64 per wave (4x4 blocking), f32 out.
// ---------------------------------------------------------------------------
__global__ __launch_bounds__(256) void k_proj(const bf16_t* __restrict__ Y,
                                              const bf16_t* __restrict__ WpT,
                                              const float* __restrict__ b_proj,
                                              float* __restrict__ out) {
  const int lane = threadIdx.x & 31;
  const int wave = threadIdx.x >> 5;
  const int NT64 = D_ / 64;                          // 16
  const int total = (B_ * S_ / 64) * NT64;           // 64*16 = 1024 waves
  int tile = blockIdx.x * 8 + wave;
  if (tile >= total) return;
  const int m0 = (tile / NT64) * 64;
  const int n0 = (tile % NT64) * 64;
  const int half = lane >> 4, l16 = lane & 15;

  f32x8 c[4][4] = {};
  for (int k0 = 0; k0 < D_; k0 += 32) {
    v16bf a[4], b[4];
#pragma unroll
    for (int i = 0; i < 4; ++i)
      a[i] = load_frag_bf16(Y, m0 + i * 16 + l16, D_, k0, half);
#pragma unroll
    for (int i = 0; i < 4; ++i)
      b[i] = load_frag_bf16(WpT, n0 + i * 16 + l16, D_, k0, half);
#pragma unroll
    for (int i = 0; i < 4; ++i)
#pragma unroll
      for (int jj = 0; jj < 4; ++jj)
        c[i][jj] = wmma_bf16(a[i], b[jj], c[i][jj]);
  }

#pragma unroll
  for (int jj = 0; jj < 4; ++jj) {
    const int gn = n0 + jj * 16 + l16;
    const float bias = b_proj[gn];
#pragma unroll
    for (int i = 0; i < 4; ++i)
#pragma unroll
      for (int j = 0; j < 8; ++j) {
        int gm = m0 + i * 16 + j + 8 * half;
        out[(size_t)gm * D_ + gn] = c[i][jj][j] + bias;
      }
  }
}

// ---------------------------------------------------------------------------
// Host launcher
// ---------------------------------------------------------------------------
extern "C" void kernel_launch(void* const* d_in, const int* in_sizes, int n_in,
                              void* d_out, int out_size, void* d_ws, size_t ws_size,
                              hipStream_t stream) {
  (void)in_sizes; (void)n_in; (void)out_size; (void)ws_size;

  const float* hidden = (const float*)d_in[0];   // [B,S,D]
  const float* amask  = (const float*)d_in[1];   // [B,1,1,S]
  const float* w_attn = (const float*)d_in[2];   // [D,3D]
  const float* b_attn = (const float*)d_in[3];   // [3D]
  const float* w_proj = (const float*)d_in[4];   // [D,D]
  const float* b_proj = (const float*)d_in[5];   // [D]

  float* out = (float*)d_out;
  float* attn_out     = out;                               // B*S*D f32
  float* attn_weights = out + (size_t)B_ * S_ * D_;        // B*H*S*S f32

  // Workspace layout (bf16 elements)
  bf16_t* ws  = (bf16_t*)d_ws;
  const size_t N_X  = (size_t)B_ * S_ * D_;     // 4194304
  const size_t N_WA = (size_t)D_ * D3_;         // 3145728
  const size_t N_WP = (size_t)D_ * D_;          // 1048576
  bf16_t* Xbf = ws;
  bf16_t* WaT = Xbf + N_X;
  bf16_t* WpT = WaT + N_WA;
  bf16_t* Qb  = WpT + N_WP;
  bf16_t* Kb  = Qb + N_X;
  bf16_t* Vt  = Kb + N_X;
  bf16_t* Yb  = Vt + N_X;

  // 1) precision conversion / weight transposes
  k_cvt_bf16<<<1024, 256, 0, stream>>>(Xbf, hidden, (int)N_X);
  k_transpose_cvt<<<2048, 256, 0, stream>>>(WaT, w_attn, D_, D3_);
  k_transpose_cvt<<<1024, 256, 0, stream>>>(WpT, w_proj, D_, D_);

  // 2) QKV projection -> Q, K, V^T (bf16), 4x4-blocked WMMA
  k_gemm_qkv<<<(64 * 48) / 8, 256, 0, stream>>>(Xbf, WaT, b_attn, Qb, Kb, Vt);

  // 3) fused scores + softmax + P@V (weights written once to d_out)
  k_attn<<<B_ * H_ * (S_ / 16), 256, 0, stream>>>(Qb, Kb, Vt, amask,
                                                  attn_weights, Yb);

  // 4) output projection -> attn_output (f32)
  k_proj<<<(64 * 16) / 8, 256, 0, stream>>>(Yb, WpT, b_proj, attn_out);
}
